// MultiHeadAttention_53901839565495
// MI455X (gfx1250) — compile-verified, matrix operand order checked by reading, not run on hardware
//
#include <hip/hip_runtime.h>
#include <hip/hip_bf16.h>

// ---------------------------------------------------------------------------
// MI455X (gfx1250) multi-head attention, WMMA f16 path.
//   B=4 S=2048 D=1024 H=16 Hd=64.  ~137 GFLOP total -> compute bound; all
//   GEMM stages go through v_wmma_f32_16x16x32_f16 (f32 accumulate).
//   Data movement: GEMM tiles staged via async global->LDS (ASYNCcnt) when
//   the toolchain exposes the builtin; flash loop prefetches next K/V block.
// ---------------------------------------------------------------------------

typedef __attribute__((ext_vector_type(16))) _Float16 v16h;
typedef __attribute__((ext_vector_type(8)))  _Float16 v8h;
typedef __attribute__((ext_vector_type(8)))  float    v8f;
typedef __attribute__((__vector_size__(16))) int      v4i;   // matches builtin param

#define AS1_ __attribute__((address_space(1)))
#define AS3_ __attribute__((address_space(3)))

#define B_  4
#define S_  2048
#define D_  1024
#define H_  16
#define HD_ 64

#if defined(__has_builtin)
#if __has_builtin(__builtin_amdgcn_global_load_async_to_lds_b128)
#define USE_ASYNC_LDS 1
#endif
#endif

__device__ __forceinline__ void wait_asynccnt0() {
#if defined(__has_builtin) && __has_builtin(__builtin_amdgcn_s_wait_asynccnt)
    __builtin_amdgcn_s_wait_asynccnt(0);
#else
    asm volatile("s_wait_asynccnt 0" ::: "memory");
#endif
}

__device__ __forceinline__ v8f v8f_zero() {
    v8f z; for (int i = 0; i < 8; ++i) z[i] = 0.0f; return z;
}

// A/B fragment gather for v_wmma_f32_16x16x32_f16 (16x32 tile, K-contiguous
// rows).  Lane L<16: row L, K={0..7,16..23}; lane L>=16: row L-16,
// K={8..15,24..31}.  Two 16-byte loads per lane.
__device__ __forceinline__ v16h frag_ld(const _Float16* p16, int stride) {
    const int lane = threadIdx.x & 31;
    const int r    = lane & 15;
    const int ko   = (lane >> 4) << 3;            // 0 or 8
    const _Float16* p = p16 + (size_t)r * stride + ko;
    v8h lo = *(const v8h*)(p);
    v8h hi = *(const v8h*)(p + 16);
    return __builtin_shufflevector(lo, hi, 0,1,2,3,4,5,6,7,8,9,10,11,12,13,14,15);
}

__device__ __forceinline__ v8f wmma32(v16h a, v16h b, v8f c) {
    return __builtin_amdgcn_wmma_f32_16x16x32_f16(
        /*neg_a=*/false, a, /*neg_b=*/false, b,
        /*c_mod=*/(short)0, c, /*reuse_a=*/false, /*reuse_b=*/false);
}

// ---------------------------------------------------------------------------
// fp32 -> f16 conversion
// ---------------------------------------------------------------------------
__global__ void cvt_f32_f16(const float* __restrict__ s, _Float16* __restrict__ d, int n) {
    int i = blockIdx.x * blockDim.x + threadIdx.x;
    if (i < n) d[i] = (_Float16)s[i];
}

// ---------------------------------------------------------------------------
// C[M,N] = A[M,K] * Bt[N,K]^T + bias[N]   (A,Bt f16 row-major, K-contiguous)
// 128x128 block, 256 threads (8 waves), each wave: 32x64 via 2x4 WMMA tiles.
// mode 0: f16 out at [B,H,S,64]        (Q / K)
// mode 1: f16 out at [B,H,64,S]        (V transposed)
// mode 2: f32 out at [M,N] row-major   (final projection)
// ---------------------------------------------------------------------------
__global__ __launch_bounds__(256)
void gemm128_wmma(const _Float16* __restrict__ A, const _Float16* __restrict__ Bt,
                  const float* __restrict__ bias, void* __restrict__ Out,
                  int M, int N, int K, int mode)
{
    __shared__ __align__(16) _Float16 sA[128 * 40];
    __shared__ __align__(16) _Float16 sB[128 * 40];

    const int m0   = blockIdx.y * 128;
    const int n0   = blockIdx.x * 128;
    const int tid  = threadIdx.x;
    const int wid  = tid >> 5;
    const int wm   = (wid & 3) * 32;              // wave m-offset in tile
    const int wn   = (wid >> 2) * 64;             // wave n-offset in tile
    const int lrow = tid >> 1;                    // 0..127 : staged row
    const int lseg = (tid & 1) * 16;              // 0/16   : 16-half segment

    v8f acc[2][4];
    for (int i = 0; i < 2; ++i)
        for (int j = 0; j < 4; ++j) acc[i][j] = v8f_zero();

    for (int k0 = 0; k0 < K; k0 += 32) {
        const _Float16* ga = A  + (size_t)(m0 + lrow) * K + k0 + lseg;
        const _Float16* gb = Bt + (size_t)(n0 + lrow) * K + k0 + lseg;
#if USE_ASYNC_LDS
        // Direct global->LDS DMA (no VGPR round-trip); tracked by ASYNCcnt.
        __builtin_amdgcn_global_load_async_to_lds_b128(
            (AS1_ v4i*)(void*)ga, (AS3_ v4i*)&sA[lrow * 40 + lseg], 0, 0);
        __builtin_amdgcn_global_load_async_to_lds_b128(
            (AS1_ v4i*)(void*)ga, (AS3_ v4i*)&sA[lrow * 40 + lseg], 16, 0);
        __builtin_amdgcn_global_load_async_to_lds_b128(
            (AS1_ v4i*)(void*)gb, (AS3_ v4i*)&sB[lrow * 40 + lseg], 0, 0);
        __builtin_amdgcn_global_load_async_to_lds_b128(
            (AS1_ v4i*)(void*)gb, (AS3_ v4i*)&sB[lrow * 40 + lseg], 16, 0);
        wait_asynccnt0();
#else
        *(v8h*)(&sA[lrow * 40 + lseg])     = *(const v8h*)(ga);
        *(v8h*)(&sA[lrow * 40 + lseg + 8]) = *(const v8h*)(ga + 8);
        *(v8h*)(&sB[lrow * 40 + lseg])     = *(const v8h*)(gb);
        *(v8h*)(&sB[lrow * 40 + lseg + 8]) = *(const v8h*)(gb + 8);
#endif
        __syncthreads();

        v16h fa0 = frag_ld(&sA[(wm +  0) * 40], 40);
        v16h fa1 = frag_ld(&sA[(wm + 16) * 40], 40);
        v16h fb0 = frag_ld(&sB[(wn +  0) * 40], 40);
        v16h fb1 = frag_ld(&sB[(wn + 16) * 40], 40);
        v16h fb2 = frag_ld(&sB[(wn + 32) * 40], 40);
        v16h fb3 = frag_ld(&sB[(wn + 48) * 40], 40);

        acc[0][0] = wmma32(fa0, fb0, acc[0][0]);
        acc[0][1] = wmma32(fa0, fb1, acc[0][1]);
        acc[0][2] = wmma32(fa0, fb2, acc[0][2]);
        acc[0][3] = wmma32(fa0, fb3, acc[0][3]);
        acc[1][0] = wmma32(fa1, fb0, acc[1][0]);
        acc[1][1] = wmma32(fa1, fb1, acc[1][1]);
        acc[1][2] = wmma32(fa1, fb2, acc[1][2]);
        acc[1][3] = wmma32(fa1, fb3, acc[1][3]);
        __syncthreads();
    }

    // epilogue: C tile lane layout -> lane holds col (lane&15), rows r+8*(lane>>4)
    const int lane  = tid & 31;
    const int cn    = lane & 15;
    const int rbase = (lane >> 4) * 8;
    for (int i = 0; i < 2; ++i)
        for (int j = 0; j < 4; ++j) {
            const int n  = n0 + wn + 16 * j + cn;
            const float bv = bias[n];
            for (int r = 0; r < 8; ++r) {
                const int m = m0 + wm + 16 * i + rbase + r;
                const float v = acc[i][j][r] + bv;
                if (mode == 2) {
                    ((float*)Out)[(size_t)m * N + n] = v;
                } else {
                    const int b = m >> 11, s = m & (S_ - 1);
                    const int h = n >> 6,  hd = n & (HD_ - 1);
                    const size_t idx = (mode == 0)
                        ? ((size_t)((b * H_ + h) * S_ + s)) * HD_ + hd
                        : ((size_t)((b * H_ + h) * HD_ + hd)) * S_ + s;
                    ((_Float16*)Out)[idx] = (_Float16)v;
                }
            }
        }
}

// ---------------------------------------------------------------------------
// Flash attention: one WG (4 waves) = 64 query rows of one (b,h).
// Each wave keeps Q[16x64] resident as two A-fragments, streams 32 keys per
// step: 4 WMMA for Q*K^T, online softmax (shfl width-16 row reductions),
// P restaged via LDS into A-fragment layout, 4 WMMA for P*V.
// Output f16 written in [B,S,D] layout so the out-projection is a plain GEMM.
// ---------------------------------------------------------------------------
__global__ __launch_bounds__(128)
void flash_attn_wmma(const _Float16* __restrict__ Q, const _Float16* __restrict__ Km,
                     const _Float16* __restrict__ Vt, const int* __restrict__ mask,
                     _Float16* __restrict__ Outv)
{
    __shared__ __align__(16) _Float16 sP[4][16 * 40];

    const int bh   = blockIdx.y;                  // b*H + h
    const int b    = bh >> 4;
    const int h    = bh & 15;
    const int wid  = threadIdx.x >> 5;
    const int lane = threadIdx.x & 31;
    const int cn   = lane & 15;
    const int rhalf = lane >> 4;                  // 0: rows 0..7, 1: rows 8..15
    const int q0   = blockIdx.x * 64 + wid * 16;

    const _Float16* Qbh = Q  + (size_t)bh * S_ * HD_;
    const _Float16* Kbh = Km + (size_t)bh * S_ * HD_;
    const _Float16* Vbh = Vt + (size_t)bh * HD_ * S_;
    const int*      mb  = mask + b * S_;
    _Float16*       myP = sP[wid];

    v16h qf0 = frag_ld(Qbh + (size_t)q0 * HD_,      HD_);
    v16h qf1 = frag_ld(Qbh + (size_t)q0 * HD_ + 32, HD_);

    v8f o[4];
    for (int j = 0; j < 4; ++j) o[j] = v8f_zero();
    float mrow[8], lrow[8];
    for (int r = 0; r < 8; ++r) { mrow[r] = -3.0e38f; lrow[r] = 0.0f; }

    const float scale = 0.125f;                   // 1/sqrt(64)

    for (int n0 = 0; n0 < S_; n0 += 32) {
        // L2/WGP prefetch of the next key/value block while this one computes.
        if (n0 + 32 < S_) {
            __builtin_prefetch(Kbh + (size_t)(n0 + 32 + cn) * HD_, 0, 3);
            __builtin_prefetch(Vbh + (size_t)(cn * 4) * S_ + n0 + 32, 0, 3);
        }

        // ---- scores S[16x32] = Q * K^T ----
        v8f s0 = v8f_zero(), s1 = v8f_zero();
        {
            v16h kf;
            kf = frag_ld(Kbh + (size_t)(n0     ) * HD_,      HD_); s0 = wmma32(qf0, kf, s0);
            kf = frag_ld(Kbh + (size_t)(n0     ) * HD_ + 32, HD_); s0 = wmma32(qf1, kf, s0);
            kf = frag_ld(Kbh + (size_t)(n0 + 16) * HD_,      HD_); s1 = wmma32(qf0, kf, s1);
            kf = frag_ld(Kbh + (size_t)(n0 + 16) * HD_ + 32, HD_); s1 = wmma32(qf1, kf, s1);
        }
        const float msk0 = mb[n0 + cn]      ? 0.0f : -1.0e30f;
        const float msk1 = mb[n0 + 16 + cn] ? 0.0f : -1.0e30f;

        // ---- online softmax ----
        for (int r = 0; r < 8; ++r) {
            float a0 = s0[r] * scale + msk0;
            float a1 = s1[r] * scale + msk1;
            float cm = fmaxf(a0, a1);
            for (int off = 8; off; off >>= 1)
                cm = fmaxf(cm, __shfl_xor(cm, off, 16));
            const float mn = fmaxf(mrow[r], cm);
            const float al = __expf(mrow[r] - mn);
            mrow[r] = mn;
            const float e0 = __expf(a0 - mn);
            const float e1 = __expf(a1 - mn);
            float ps = e0 + e1;
            for (int off = 8; off; off >>= 1)
                ps += __shfl_xor(ps, off, 16);
            lrow[r] = lrow[r] * al + ps;
            for (int j = 0; j < 4; ++j) o[j][r] *= al;
            const int row = r + 8 * rhalf;
            myP[row * 40 + cn]      = (_Float16)e0;
            myP[row * 40 + 16 + cn] = (_Float16)e1;
        }
        asm volatile("s_wait_dscnt 0" ::: "memory");  // P staged in LDS (per-wave)

        // ---- O += P * V ----
        v16h pf = frag_ld(myP, 40);
        for (int j = 0; j < 4; ++j) {
            v16h vf = frag_ld(Vbh + (size_t)(j * 16) * S_ + n0, S_);
            o[j] = wmma32(pf, vf, o[j]);
        }
    }

    // ---- finalize + write [B,S,D] f16 ----
    for (int j = 0; j < 4; ++j)
        for (int r = 0; r < 8; ++r) {
            const float v = o[j][r] / lrow[r];
            const int srow = q0 + r + 8 * rhalf;
            const size_t idx = ((size_t)(b * S_ + srow)) * D_ + h * HD_ + j * 16 + cn;
            Outv[idx] = (_Float16)v;
        }
}

// ---------------------------------------------------------------------------
extern "C" void kernel_launch(void* const* d_in, const int* in_sizes, int n_in,
                              void* d_out, int out_size, void* d_ws, size_t ws_size,
                              hipStream_t stream) {
    const float* x    = (const float*)d_in[0];
    const int*   mask = (const int*)  d_in[1];
    const float* Wq   = (const float*)d_in[2];
    const float* bq   = (const float*)d_in[3];
    const float* Wk   = (const float*)d_in[4];
    const float* bk   = (const float*)d_in[5];
    const float* Wv   = (const float*)d_in[6];
    const float* bv   = (const float*)d_in[7];
    const float* Wo   = (const float*)d_in[8];
    const float* bo   = (const float*)d_in[9];

    const int MT = B_ * S_;              // 8192 rows
    const int NX = MT * D_;              // 8388608
    const int NW = D_ * D_;              // 1048576

    _Float16* ws    = (_Float16*)d_ws;
    _Float16* xh    = ws;                // [8192,1024]
    _Float16* wqh   = xh   + NX;
    _Float16* wkh   = wqh  + NW;
    _Float16* wvh   = wkh  + NW;
    _Float16* woh   = wvh  + NW;
    _Float16* qh    = woh  + NW;         // [B,H,S,64]
    _Float16* kh    = qh   + NX;         // [B,H,S,64]
    _Float16* vth   = kh   + NX;         // [B,H,64,S]
    _Float16* valsh = vth  + NX;         // [B,S,D]

    cvt_f32_f16<<<(NX + 255) / 256, 256, 0, stream>>>(x,  xh,  NX);
    cvt_f32_f16<<<(NW + 255) / 256, 256, 0, stream>>>(Wq, wqh, NW);
    cvt_f32_f16<<<(NW + 255) / 256, 256, 0, stream>>>(Wk, wkh, NW);
    cvt_f32_f16<<<(NW + 255) / 256, 256, 0, stream>>>(Wv, wvh, NW);
    cvt_f32_f16<<<(NW + 255) / 256, 256, 0, stream>>>(Wo, woh, NW);

    dim3 gg(D_ / 128, MT / 128);         // (8, 64)
    gemm128_wmma<<<gg, 256, 0, stream>>>(xh, wqh, bq, qh,  MT, D_, D_, 0);
    gemm128_wmma<<<gg, 256, 0, stream>>>(xh, wkh, bk, kh,  MT, D_, D_, 0);
    gemm128_wmma<<<gg, 256, 0, stream>>>(xh, wvh, bv, vth, MT, D_, D_, 1);

    flash_attn_wmma<<<dim3(S_ / 64, B_ * H_), 128, 0, stream>>>(qh, kh, vth, mask, valsh);

    gemm128_wmma<<<gg, 256, 0, stream>>>(valsh, woh, bo, d_out, MT, D_, D_, 2);
}
